// SelectiveLinearAttention_4037269258996
// MI455X (gfx1250) — compile-verified, matrix operand order checked by reading, not run on hardware
//
#include <hip/hip_runtime.h>
#include <hip/hip_bf16.h>

typedef _Float16 v8h  __attribute__((ext_vector_type(8)));
typedef _Float16 v16h __attribute__((ext_vector_type(16)));
typedef float    v8f  __attribute__((ext_vector_type(8)));

#define HID   1024
#define SEQ   2048
#define BB    2
#define NHD   16
#define HDIM  64
#define MTOK  (BB*SEQ)      /* 4096 tokens */
#define QSCALE 0.125f       /* HD^-0.5 */

// Load a 16x32 f16 A/B fragment half for this lane: elements [0..7] = p[0..7],
// elements [8..15] = p[16..23]  (K-halves per CDNA5 16-bit WMMA VGPR layout).
__device__ __forceinline__ v16h load_frag(const _Float16* __restrict__ p) {
  v8h lo = *(const v8h*)(p);
  v8h hi = *(const v8h*)(p + 16);
  v16h r;
#pragma unroll
  for (int i = 0; i < 8; ++i) { r[i] = lo[i]; r[i + 8] = hi[i]; }
  return r;
}

__device__ __forceinline__ v8f wmma16(v16h a, v16h b, v8f c) {
  return __builtin_amdgcn_wmma_f32_16x16x32_f16(false, a, false, b, (short)0, c,
                                                false, false);
}

// ---------------------------------------------------------------------------
// Fused triple LayerNorm: x[4096,1024] f32 -> lnq/lnk/lnv f16
// ---------------------------------------------------------------------------
__global__ void ln3_kernel(const float* __restrict__ x,
                           const float* __restrict__ qw, const float* __restrict__ qb,
                           const float* __restrict__ kw, const float* __restrict__ kb,
                           const float* __restrict__ vw, const float* __restrict__ vb,
                           _Float16* __restrict__ lnq,
                           _Float16* __restrict__ lnk,
                           _Float16* __restrict__ lnv) {
  __shared__ float s1[256];
  __shared__ float s2[256];
  const int row = blockIdx.x;
  const int tid = threadIdx.x;
  const float4 v = ((const float4*)(x + (size_t)row * HID))[tid];
  float sum = v.x + v.y + v.z + v.w;
  float sq  = v.x * v.x + v.y * v.y + v.z * v.z + v.w * v.w;
  s1[tid] = sum; s2[tid] = sq;
  __syncthreads();
#pragma unroll
  for (int o = 128; o > 0; o >>= 1) {
    if (tid < o) { s1[tid] += s1[tid + o]; s2[tid] += s2[tid + o]; }
    __syncthreads();
  }
  const float mu   = s1[0] * (1.0f / HID);
  const float var  = s2[0] * (1.0f / HID) - mu * mu;
  const float rstd = rsqrtf(var + 1e-5f);
  const int c0 = tid * 4;
  const float e[4] = {v.x, v.y, v.z, v.w};
#pragma unroll
  for (int i = 0; i < 4; ++i) {
    const int c = c0 + i;
    const float nv = (e[i] - mu) * rstd;
    const size_t o = (size_t)row * HID + c;
    lnq[o] = (_Float16)(nv * qw[c] + qb[c]);
    lnk[o] = (_Float16)(nv * kw[c] + kb[c]);
    lnv[o] = (_Float16)(nv * vw[c] + vb[c]);
  }
}

// ---------------------------------------------------------------------------
// f32 -> f16 weight conversion
// ---------------------------------------------------------------------------
__global__ void cvt_f32_f16(const float* __restrict__ src,
                            _Float16* __restrict__ dst, int n) {
  const int i = blockIdx.x * blockDim.x + threadIdx.x;
  if (i < n) dst[i] = (_Float16)src[i];
}

// ---------------------------------------------------------------------------
// WMMA GEMM: C[M=4096, N=1024] = A[M,K=1024] @ W[N,K]^T + bias
// One wave computes a 32(M) x 64(N) tile (8 accumulators). The K loop (32
// steps of K=32) is FULLY UNROLLED: no loop-carried phis, accumulators stay
// pinned in one register set, and the pressure-aware scheduler hoists each
// step's 12 global_load_b128 across the previous step's 8 WMMAs (partial
// s_wait_loadcnt), with no rotation/phi copy moves.
// __launch_bounds__(256,1): allow the allocator the full VGPR budget.
// mode 0: q  -> f16 [b,h,s,d], (acc+bias)*QSCALE
// mode 1: k  -> f16 [b,h,s,d]
// mode 2: v  -> f16 [b,h,d,s]   (transposed so attention P@V streams contiguously)
// mode 3: o  -> f32 row-major [m, n] (final output)
// ---------------------------------------------------------------------------
__global__ void __launch_bounds__(256, 1)
wmma_gemm(const _Float16* __restrict__ A,
          const _Float16* __restrict__ W,
          const float* __restrict__ bias,
          _Float16* __restrict__ outH,
          float* __restrict__ outF,
          int mode) {
  const int wid   = (blockIdx.x * blockDim.x + threadIdx.x) >> 5;
  const int lane  = threadIdx.x & 31;
  const int mtile = wid >> 4;          // 0..127 (32-row blocks)
  const int ntile = wid & 15;          // 0..15  (64-col blocks)
  const int m0 = mtile * 32;
  const int n0 = ntile * 64;
  const int mr  = lane & 15;           // M (for A) / N (for B) within tile
  const int kh8 = (lane >> 4) << 3;    // K-half select: 0 or 8

  v8f acc[8] = {v8f{}, v8f{}, v8f{}, v8f{}, v8f{}, v8f{}, v8f{}, v8f{}};
  const _Float16* pa0 = A + (size_t)(m0 + mr) * HID + kh8;
  const _Float16* pa1 = A + (size_t)(m0 + 16 + mr) * HID + kh8;
  const _Float16* pb  = W + (size_t)(n0 + mr) * HID + kh8;

#pragma unroll
  for (int k0 = 0; k0 < HID; k0 += 32) {
    const v16h a0 = load_frag(pa0 + k0);
    const v16h a1 = load_frag(pa1 + k0);
    const v16h b0 = load_frag(pb + k0);
    const v16h b1 = load_frag(pb + 16 * HID + k0);
    const v16h b2 = load_frag(pb + 32 * HID + k0);
    const v16h b3 = load_frag(pb + 48 * HID + k0);
    acc[0] = wmma16(a0, b0, acc[0]);
    acc[1] = wmma16(a0, b1, acc[1]);
    acc[2] = wmma16(a0, b2, acc[2]);
    acc[3] = wmma16(a0, b3, acc[3]);
    acc[4] = wmma16(a1, b0, acc[4]);
    acc[5] = wmma16(a1, b1, acc[5]);
    acc[6] = wmma16(a1, b2, acc[6]);
    acc[7] = wmma16(a1, b3, acc[7]);
  }

  const int hsel = lane >> 4;
#pragma unroll
  for (int half = 0; half < 2; ++half) {
#pragma unroll
    for (int j = 0; j < 4; ++j) {
      const int n = n0 + j * 16 + mr;
      const float bn = bias[n];
#pragma unroll
      for (int r = 0; r < 8; ++r) {
        const int m = m0 + half * 16 + r + 8 * hsel;   // token index
        float val = acc[half * 4 + j][r] + bn;
        if (mode == 0) val *= QSCALE;
        const int b = m >> 11, s = m & (SEQ - 1);
        const int h = n >> 6,  d = n & (HDIM - 1);
        if (mode <= 1) {
          outH[(((size_t)b * NHD + h) * SEQ + s) * HDIM + d] = (_Float16)val;
        } else if (mode == 2) {
          outH[(((size_t)b * NHD + h) * HDIM + d) * SEQ + s] = (_Float16)val;
        } else {
          outF[(size_t)m * HID + n] = val;
        }
      }
    }
  }
}

// ---------------------------------------------------------------------------
// Flash attention: one wave per 16-row q tile of one (b,h).
// q,k f16 [b,h,s,d]; vT f16 [b,h,d,s]; out f16 [b, s, h*64+d] (== [B,S,H]).
// K-fragments are rotated with a branch-free wrap prefetch; "#pragma unroll 2"
// turns the rotation into an automatic register ping-pong (no copy moves).
// V loads are issued ahead of the softmax / LDS barrier for overlap.
// ---------------------------------------------------------------------------
__global__ void __launch_bounds__(256, 1)
attn_kernel(const _Float16* __restrict__ qg,
            const _Float16* __restrict__ kg,
            const _Float16* __restrict__ vtg,
            _Float16* __restrict__ og) {
  __shared__ __align__(16) _Float16 plds[8][16 * 32];  // P staging per wave
  const int wave = threadIdx.x >> 5;
  const int lane = threadIdx.x & 31;
  const int bh = blockIdx.x;                  // 0..31 : b*NH + h
  const int qt = blockIdx.y * 8 + wave;       // 0..127
  const int mr  = lane & 15;
  const int kh8 = (lane >> 4) << 3;
  const int hsel = lane >> 4;

  const _Float16* qp = qg  + (size_t)bh * SEQ * HDIM;
  const _Float16* kp = kg  + (size_t)bh * SEQ * HDIM;
  const _Float16* vp = vtg + (size_t)bh * HDIM * SEQ;
  _Float16* lp = &plds[wave][0];

  // Q fragments (16 rows x 64 d) : two K=32 A-fragments, reused for all keys
  const v16h aq0 = load_frag(qp + (size_t)(qt * 16 + mr) * HDIM + 0  + kh8);
  const v16h aq1 = load_frag(qp + (size_t)(qt * 16 + mr) * HDIM + 32 + kh8);

  float mrow[8], lrow[8];
  v8f oacc[4] = {v8f{}, v8f{}, v8f{}, v8f{}};
#pragma unroll
  for (int r = 0; r < 8; ++r) { mrow[r] = -1.0e30f; lrow[r] = 0.0f; }

  // prologue: K fragments for chunk 0
  v16h kb0 = load_frag(kp + (size_t)(0 + mr) * HDIM + 0  + kh8);
  v16h kb1 = load_frag(kp + (size_t)(0 + mr) * HDIM + 32 + kh8);
  v16h kb2 = load_frag(kp + (size_t)(16 + mr) * HDIM + 0  + kh8);
  v16h kb3 = load_frag(kp + (size_t)(16 + mr) * HDIM + 32 + kh8);

#pragma unroll 2
  for (int kc = 0; kc < SEQ; kc += 32) {
    // ---- scores: two 16x16 tiles over keys [kc, kc+32)
    v8f s0 = {}; s0 = wmma16(aq0, kb0, s0); s0 = wmma16(aq1, kb1, s0);
    v8f s1 = {}; s1 = wmma16(aq0, kb2, s1); s1 = wmma16(aq1, kb3, s1);

    // ---- prefetch next chunk's K fragments (wrap to 0 on last iteration:
    //      branch-free; one redundant 4 KB reload, never consumed)
    const int kn = (kc + 32 < SEQ) ? (kc + 32) : 0;
    kb0 = load_frag(kp + (size_t)(kn + mr) * HDIM + 0  + kh8);
    kb1 = load_frag(kp + (size_t)(kn + mr) * HDIM + 32 + kh8);
    kb2 = load_frag(kp + (size_t)(kn + 16 + mr) * HDIM + 0  + kh8);
    kb3 = load_frag(kp + (size_t)(kn + 16 + mr) * HDIM + 32 + kh8);

    // ---- V fragments for the current chunk: issue before the softmax and
    //      before the LDS compiler barrier so they overlap the VALU work
    const v16h vb0 = load_frag(vp + (size_t)(0  + mr) * SEQ + kc + kh8);
    const v16h vb1 = load_frag(vp + (size_t)(16 + mr) * SEQ + kc + kh8);
    const v16h vb2 = load_frag(vp + (size_t)(32 + mr) * SEQ + kc + kh8);
    const v16h vb3 = load_frag(vp + (size_t)(48 + mr) * SEQ + kc + kh8);

    // ---- online softmax (rows live in one half-wave -> width-16 shuffles)
#pragma unroll
    for (int r = 0; r < 8; ++r) {
      float mx = fmaxf(s0[r], s1[r]);
#pragma unroll
      for (int o = 8; o > 0; o >>= 1) mx = fmaxf(mx, __shfl_xor(mx, o, 16));
      const float newm  = fmaxf(mrow[r], mx);
      const float alpha = __expf(mrow[r] - newm);
      const float p0 = __expf(s0[r] - newm);
      const float p1 = __expf(s1[r] - newm);
      float rs = p0 + p1;
#pragma unroll
      for (int o = 8; o > 0; o >>= 1) rs += __shfl_xor(rs, o, 16);
      lrow[r] = lrow[r] * alpha + rs;
      mrow[r] = newm;
#pragma unroll
      for (int j = 0; j < 4; ++j) oacc[j][r] *= alpha;
      const int row = r + 8 * hsel;
      lp[row * 32 + mr]      = (_Float16)p0;
      lp[row * 32 + 16 + mr] = (_Float16)p1;
    }
    asm volatile("s_wait_dscnt 0" ::: "memory");

    // ---- P (D-layout) -> A-fragment via LDS, then P @ V
    v16h ap;
    {
      const v8h lo = *(const v8h*)(lp + mr * 32 + kh8);
      const v8h hi = *(const v8h*)(lp + mr * 32 + 16 + kh8);
#pragma unroll
      for (int i = 0; i < 8; ++i) { ap[i] = lo[i]; ap[i + 8] = hi[i]; }
    }
    oacc[0] = wmma16(ap, vb0, oacc[0]);
    oacc[1] = wmma16(ap, vb1, oacc[1]);
    oacc[2] = wmma16(ap, vb2, oacc[2]);
    oacc[3] = wmma16(ap, vb3, oacc[3]);
  }

  // ---- normalize + store to [B, S, H] as f16
  const int b = bh >> 4, h = bh & 15;
#pragma unroll
  for (int j = 0; j < 4; ++j) {
#pragma unroll
    for (int r = 0; r < 8; ++r) {
      const int s = qt * 16 + r + 8 * hsel;
      const int d = j * 16 + mr;
      const float val = oacc[j][r] / lrow[r];
      og[((size_t)b * SEQ + s) * HID + h * HDIM + d] = (_Float16)val;
    }
  }
}

// ---------------------------------------------------------------------------
extern "C" void kernel_launch(void* const* d_in, const int* in_sizes, int n_in,
                              void* d_out, int out_size, void* d_ws, size_t ws_size,
                              hipStream_t stream) {
  const float* x    = (const float*)d_in[0];
  const float* lnqw = (const float*)d_in[1];
  const float* lnqb = (const float*)d_in[2];
  const float* lnkw = (const float*)d_in[3];
  const float* lnkb = (const float*)d_in[4];
  const float* lnvw = (const float*)d_in[5];
  const float* lnvb = (const float*)d_in[6];
  const float* wq   = (const float*)d_in[7];
  const float* bq   = (const float*)d_in[8];
  const float* wk   = (const float*)d_in[9];
  const float* bk   = (const float*)d_in[10];
  const float* wv   = (const float*)d_in[11];
  const float* bv   = (const float*)d_in[12];
  const float* wo   = (const float*)d_in[13];
  const float* bo   = (const float*)d_in[14];
  float* out = (float*)d_out;

  const size_t MB = (size_t)1 << 20;
  uint8_t* w = (uint8_t*)d_ws;
  _Float16* lnq = (_Float16*)(w + 0 * MB);    // 8 MB each
  _Float16* lnk = (_Float16*)(w + 8 * MB);
  _Float16* lnv = (_Float16*)(w + 16 * MB);
  _Float16* wqh = (_Float16*)(w + 24 * MB);   // 2 MB each
  _Float16* wkh = (_Float16*)(w + 26 * MB);
  _Float16* wvh = (_Float16*)(w + 28 * MB);
  _Float16* woh = (_Float16*)(w + 30 * MB);
  _Float16* qh  = (_Float16*)(w + 32 * MB);   // 8 MB each
  _Float16* kh  = (_Float16*)(w + 40 * MB);
  _Float16* vth = (_Float16*)(w + 48 * MB);
  _Float16* oh  = (_Float16*)(w + 56 * MB);

  // 1. fused triple LayerNorm -> f16 activations
  ln3_kernel<<<MTOK, 256, 0, stream>>>(x, lnqw, lnqb, lnkw, lnkb, lnvw, lnvb,
                                       lnq, lnk, lnv);
  // 2. weight conversions f32 -> f16
  const int NW = HID * HID;
  cvt_f32_f16<<<NW / 256, 256, 0, stream>>>(wq, wqh, NW);
  cvt_f32_f16<<<NW / 256, 256, 0, stream>>>(wk, wkh, NW);
  cvt_f32_f16<<<NW / 256, 256, 0, stream>>>(wv, wvh, NW);
  cvt_f32_f16<<<NW / 256, 256, 0, stream>>>(wo, woh, NW);
  // 3. QKV projections (2048 waves; wave = 32x64 tile)
  wmma_gemm<<<256, 256, 0, stream>>>(lnq, wqh, bq, qh, nullptr, 0);
  wmma_gemm<<<256, 256, 0, stream>>>(lnk, wkh, bk, kh, nullptr, 1);
  wmma_gemm<<<256, 256, 0, stream>>>(lnv, wvh, bv, vth, nullptr, 2);
  // 4. flash attention: 32 (b,h) x 128 q-tiles / 8 waves per block
  attn_kernel<<<dim3(BB * NHD, 16), 256, 0, stream>>>(qh, kh, vth, oh);
  // 5. output projection -> f32 d_out
  wmma_gemm<<<256, 256, 0, stream>>>(oh, woh, bo, nullptr, out, 3);
}
